// TimeSeriesEmbedding_5111011082246
// MI455X (gfx1250) — compile-verified, matrix operand order checked by reading, not run on hardware
//
#include <hip/hip_runtime.h>
#include <hip/hip_bf16.h>
#include <cstdint>
#include <cstddef>

// Problem constants (match the reference)
#define B_   256
#define T_   8192
#define P_   16
#define H_   1024
#define TF_  (T_ * 2)   // floats per batch row of x (val, mask interleaved)

#define MT_  32         // M rows (patches) per workgroup: 2 WMMA row-tiles

typedef __attribute__((ext_vector_type(16))) __bf16 v16bf;
typedef __attribute__((ext_vector_type(8)))  __bf16 v8bf;
typedef __attribute__((ext_vector_type(8)))  float  v8f;

__device__ __forceinline__ float gelu_exact(float x) {
    return 0.5f * x * (1.0f + erff(x * 0.70710678118654752440f));
}

// Build a 16-wide bf16 WMMA fragment from two contiguous 8-element (16B) chunks.
__device__ __forceinline__ v16bf ld_frag(const __bf16* p0, const __bf16* p1) {
    v8bf lo = *(const v8bf*)p0;
    v8bf hi = *(const v8bf*)p1;
    return __builtin_shufflevector(lo, hi, 0, 1, 2, 3, 4, 5, 6, 7,
                                           8, 9, 10, 11, 12, 13, 14, 15);
}

// ---------------------------------------------------------------------------
// Kernel 1: per-batch mask reduction -> valid, pc ; also write pc to d_out tail
// ---------------------------------------------------------------------------
__global__ void k_valid_pc(const float* __restrict__ x,
                           int* __restrict__ valid,
                           int* __restrict__ pc,
                           int* __restrict__ pc_out) {
    __shared__ float red[256];
    const int b = blockIdx.x;
    const int t = threadIdx.x;
    const float* row = x + (size_t)b * TF_;
    float s = 0.0f;
    for (int i = t; i < T_; i += 256) s += row[i * 2 + 1];
    red[t] = s;
    __syncthreads();
    for (int off = 128; off > 0; off >>= 1) {
        if (t < off) red[t] += red[t + off];
        __syncthreads();
    }
    if (t == 0) {
        int v = (int)(red[0] + 0.5f);
        valid[b] = v;
        int c = (v + P_ - 1) / P_;
        pc[b] = c;
        pc_out[b] = c;
    }
}

// ---------------------------------------------------------------------------
// Kernel 2: exclusive scan of pc (256 entries, single block)
// ---------------------------------------------------------------------------
__global__ void k_scan(const int* __restrict__ pc, int* __restrict__ starts) {
    __shared__ int sm[256];
    const int t = threadIdx.x;
    int v = pc[t];
    sm[t] = v;
    __syncthreads();
    for (int off = 1; off < 256; off <<= 1) {
        int add = (t >= off) ? sm[t - off] : 0;
        __syncthreads();
        sm[t] += add;
        __syncthreads();
    }
    starts[t] = sm[t] - v;
}

// ---------------------------------------------------------------------------
// Kernel 3: patch index -> (batch, patch-in-batch) map
// ---------------------------------------------------------------------------
__global__ void k_fill_map(const int* __restrict__ pc,
                           const int* __restrict__ starts,
                           int* __restrict__ patch_b,
                           int* __restrict__ patch_p) {
    const int b = blockIdx.x;
    const int s = starts[b];
    const int n = pc[b];
    for (int p = threadIdx.x; p < n; p += blockDim.x) {
        patch_b[s + p] = b;
        patch_p[s + p] = p;
    }
}

// ---------------------------------------------------------------------------
// Kernel 4: weights -> bf16, N-major (transposed).  w1 padded K 16 -> 32.
// ---------------------------------------------------------------------------
__global__ void k_conv_w(const float* __restrict__ w1,
                         const float* __restrict__ w2,
                         const float* __restrict__ w3,
                         __bf16* __restrict__ w1t,   // [H][32]
                         __bf16* __restrict__ w2t,   // [H][H]
                         __bf16* __restrict__ w3t) { // [H][H]
    const int stride = gridDim.x * blockDim.x;
    const int i0 = blockIdx.x * blockDim.x + threadIdx.x;
    for (int i = i0; i < H_ * 32; i += stride) {
        int n = i >> 5, k = i & 31;
        w1t[i] = (k < P_) ? (__bf16)w1[k * H_ + n] : (__bf16)0.0f;
    }
    for (int j = i0; j < H_ * H_; j += stride) {
        int n = j >> 10, k = j & (H_ - 1);
        w2t[j] = (__bf16)w2[k * H_ + n];
        w3t[j] = (__bf16)w3[k * H_ + n];
    }
}

// ---------------------------------------------------------------------------
// Kernel 5: fused gather + 3-layer MLP, 32 patches (2 WMMA row-tiles) per WG.
//   8 wave32 per WG; each wave owns 128 output columns (8 WMMA N-tiles) and
//   BOTH row-tiles, so every B-fragment load feeds 2 WMMAs (32 FLOP/L2-byte).
//   hA/hB: 32 x 1024 bf16 activation tiles in LDS (128 KB total -- needs the
//   CDNA5 320KB WGP LDS).  Patch tile (32x32 bf16) aliases the front of hB.
// ---------------------------------------------------------------------------
__launch_bounds__(256)
__global__ void k_mlp(const float* __restrict__ x,
                      const int* __restrict__ valid,
                      const int* __restrict__ patch_b,
                      const int* __restrict__ patch_p,
                      const __bf16* __restrict__ w1t,
                      const __bf16* __restrict__ w2t,
                      const __bf16* __restrict__ w3t,
                      const float* __restrict__ b1,
                      const float* __restrict__ b2,
                      const float* __restrict__ b3,
                      float* __restrict__ out,
                      int total_patches) {
    __shared__ __bf16 hA[MT_ * H_];
    __shared__ __bf16 hB[MT_ * H_];

    const int tid  = threadIdx.x;
    const int lane = tid & 31;
    const int wave = tid >> 5;
    const int tile = blockIdx.x;

    // ---- stage 0: gather the 32x16 patch tile into LDS, zero-pad K to 32 ----
    for (int e = tid; e < MT_ * 16; e += 256) {
        int r = e >> 4;          // 0..31 patch row in tile
        int k = e & 15;          // 0..15 element in patch
        int j = tile * MT_ + r;  // global patch index
        float v = 0.0f;
        if (j < total_patches) {
            int b   = patch_b[j];
            int p   = patch_p[j];
            int idx = p * P_ + k;
            int vb  = valid[b];
            int ci  = idx < (T_ - 1) ? idx : (T_ - 1);
            float val = x[(size_t)b * TF_ + (size_t)ci * 2];
            v = (idx < vb) ? val : 0.0f;
        }
        hB[r * 32 + k]      = (__bf16)v;
        hB[r * 32 + 16 + k] = (__bf16)0.0f;
    }
    __syncthreads();

    // WMMA lane mapping (16-bit A/B fragments, wave32):
    //   lanes 0-15: row/col = lane,    K sub-block 0 (k = 0..7  then 16..23)
    //   lanes16-31: row/col = lane-16, K sub-block 1 (k = 8..15 then 24..31)
    const int rsel  = lane & 15;
    const int kgrp  = (lane >> 4) * 8;
    const int rbase = (lane >> 4) * 8;   // C/D: vgpr i -> row i + rbase
    const int cb    = wave * 128;        // this wave's N base

    v8f acc0[8], acc1[8];                // row-tile 0 / row-tile 1 accumulators
    const v8f vzero = {};

    // ---- GEMM1: patch(32x32) @ w1 -> hA = gelu(. + b1), 1 WMMA pair per tile
    {
        v16bf af0 = ld_frag(&hB[rsel * 32 + kgrp],
                            &hB[rsel * 32 + 16 + kgrp]);
        v16bf af1 = ld_frag(&hB[(16 + rsel) * 32 + kgrp],
                            &hB[(16 + rsel) * 32 + 16 + kgrp]);
#pragma unroll
        for (int nt = 0; nt < 8; ++nt) {
            int col = cb + nt * 16 + rsel;
            const __bf16* bp = &w1t[(size_t)col * 32 + kgrp];
            v16bf bfrag = ld_frag(bp, bp + 16);
            acc0[nt] = __builtin_amdgcn_wmma_f32_16x16x32_bf16(
                false, af0, false, bfrag, (short)0, vzero, false, false);
            acc1[nt] = __builtin_amdgcn_wmma_f32_16x16x32_bf16(
                false, af1, false, bfrag, (short)0, vzero, false, false);
        }
#pragma unroll
        for (int nt = 0; nt < 8; ++nt) {
            int col = cb + nt * 16 + rsel;
            float bias = b1[col];
#pragma unroll
            for (int i = 0; i < 8; ++i) {
                hA[(i + rbase) * H_ + col]      = (__bf16)gelu_exact(acc0[nt][i] + bias);
                hA[(16 + i + rbase) * H_ + col] = (__bf16)gelu_exact(acc1[nt][i] + bias);
            }
        }
    }
    __syncthreads();

    // ---- GEMM2: hA(32x1024) @ w2 -> hB = gelu(. + b2) ----
#pragma unroll
    for (int nt = 0; nt < 8; ++nt) { acc0[nt] = vzero; acc1[nt] = vzero; }
    for (int kb = 0; kb < H_; kb += 32) {
        v16bf af0 = ld_frag(&hA[rsel * H_ + kb + kgrp],
                            &hA[rsel * H_ + kb + 16 + kgrp]);
        v16bf af1 = ld_frag(&hA[(16 + rsel) * H_ + kb + kgrp],
                            &hA[(16 + rsel) * H_ + kb + 16 + kgrp]);
#pragma unroll
        for (int nt = 0; nt < 8; ++nt) {
            int col = cb + nt * 16 + rsel;
            const __bf16* bp = &w2t[(size_t)col * H_ + kb + kgrp];
            v16bf bfrag = ld_frag(bp, bp + 16);
            acc0[nt] = __builtin_amdgcn_wmma_f32_16x16x32_bf16(
                false, af0, false, bfrag, (short)0, acc0[nt], false, false);
            acc1[nt] = __builtin_amdgcn_wmma_f32_16x16x32_bf16(
                false, af1, false, bfrag, (short)0, acc1[nt], false, false);
        }
    }
#pragma unroll
    for (int nt = 0; nt < 8; ++nt) {
        int col = cb + nt * 16 + rsel;
        float bias = b2[col];
#pragma unroll
        for (int i = 0; i < 8; ++i) {
            hB[(i + rbase) * H_ + col]      = (__bf16)gelu_exact(acc0[nt][i] + bias);
            hB[(16 + i + rbase) * H_ + col] = (__bf16)gelu_exact(acc1[nt][i] + bias);
        }
    }
    __syncthreads();

    // ---- GEMM3: hB(32x1024) @ w3 + b3 -> out (f32) ----
#pragma unroll
    for (int nt = 0; nt < 8; ++nt) { acc0[nt] = vzero; acc1[nt] = vzero; }
    for (int kb = 0; kb < H_; kb += 32) {
        v16bf af0 = ld_frag(&hB[rsel * H_ + kb + kgrp],
                            &hB[rsel * H_ + kb + 16 + kgrp]);
        v16bf af1 = ld_frag(&hB[(16 + rsel) * H_ + kb + kgrp],
                            &hB[(16 + rsel) * H_ + kb + 16 + kgrp]);
#pragma unroll
        for (int nt = 0; nt < 8; ++nt) {
            int col = cb + nt * 16 + rsel;
            const __bf16* bp = &w3t[(size_t)col * H_ + kb + kgrp];
            v16bf bfrag = ld_frag(bp, bp + 16);
            acc0[nt] = __builtin_amdgcn_wmma_f32_16x16x32_bf16(
                false, af0, false, bfrag, (short)0, acc0[nt], false, false);
            acc1[nt] = __builtin_amdgcn_wmma_f32_16x16x32_bf16(
                false, af1, false, bfrag, (short)0, acc1[nt], false, false);
        }
    }
#pragma unroll
    for (int nt = 0; nt < 8; ++nt) {
        int col = cb + nt * 16 + rsel;
        float bias = b3[col];
#pragma unroll
        for (int i = 0; i < 8; ++i) {
            int j0 = tile * MT_ + (i + rbase);
            int j1 = tile * MT_ + (16 + i + rbase);
            if (j0 < total_patches) out[(size_t)j0 * H_ + col] = acc0[nt][i] + bias;
            if (j1 < total_patches) out[(size_t)j1 * H_ + col] = acc1[nt][i] + bias;
        }
    }
}

// ---------------------------------------------------------------------------
// Host-side launcher
// ---------------------------------------------------------------------------
extern "C" void kernel_launch(void* const* d_in, const int* in_sizes, int n_in,
                              void* d_out, int out_size, void* d_ws, size_t ws_size,
                              hipStream_t stream) {
    const float* x  = (const float*)d_in[0];
    const float* w1 = (const float*)d_in[1];
    const float* b1 = (const float*)d_in[2];
    const float* w2 = (const float*)d_in[3];
    const float* b2 = (const float*)d_in[4];
    const float* w3 = (const float*)d_in[5];
    const float* b3 = (const float*)d_in[6];
    // d_in[7] is total_patches on device; derive it host-side from out_size:
    // out_size = total_patches * H + B   (out f32 block + pc int32 tail)
    const int total_patches = (out_size - B_) / H_;

    // Workspace layout (ints are 4B, bf16 are 2B); max patches = 256*512 = 131072
    char* ws = (char*)d_ws;
    int*    valid   = (int*)(ws + 0);
    int*    pc      = (int*)(ws + 1024);
    int*    starts  = (int*)(ws + 2048);
    int*    patch_b = (int*)(ws + 4096);
    int*    patch_p = (int*)(ws + 4096 + 131072 * 4);
    __bf16* w1t     = (__bf16*)(ws + 4096 + 2 * 131072 * 4);
    __bf16* w2t     = (__bf16*)((char*)w1t + (size_t)H_ * 32 * 2);
    __bf16* w3t     = (__bf16*)((char*)w2t + (size_t)H_ * H_ * 2);

    float* out   = (float*)d_out;
    int*   pcout = (int*)d_out + (size_t)total_patches * H_;

    k_valid_pc<<<B_, 256, 0, stream>>>(x, valid, pc, pcout);
    k_scan<<<1, 256, 0, stream>>>(pc, starts);
    k_fill_map<<<B_, 256, 0, stream>>>(pc, starts, patch_b, patch_p);
    k_conv_w<<<2048, 256, 0, stream>>>(w1, w2, w3, w1t, w2t, w3t);

    const int num_tiles = (total_patches + MT_ - 1) / MT_;
    k_mlp<<<num_tiles, 256, 0, stream>>>(x, valid, patch_b, patch_p,
                                         w1t, w2t, w3t, b1, b2, b3,
                                         out, total_patches);
}
